// CROSSLoss_13975823581573
// MI455X (gfx1250) — matrix operand order repaired and checked by reading, not compiled, not run on hardware
//
#include <hip/hip_runtime.h>
#include <stdint.h>

#define BB   8192
#define DD   1024
#define CC   1000
#define CPAD 1024

typedef __attribute__((ext_vector_type(16))) __bf16 bf16x16;
typedef __attribute__((ext_vector_type(8)))  __bf16 bf16x8;
typedef __attribute__((ext_vector_type(8)))  float  f32x8;

__device__ __forceinline__ float wsum(float v) {
#pragma unroll
  for (int o = 16; o > 0; o >>= 1) v += __shfl_xor(v, o, 32);
  return v;
}
__device__ __forceinline__ float wmaxr(float v) {
#pragma unroll
  for (int o = 16; o > 0; o >>= 1) v = fmaxf(v, __shfl_xor(v, o, 32));
  return v;
}

// ---------------- 1) reciprocal row norms: rnorm[i] = 1/max(||x_i||, eps) ----
__global__ void k_rnorm(const float* __restrict__ x, float* __restrict__ rnorm) {
  const int wave = threadIdx.x >> 5, lane = threadIdx.x & 31;
  const int row  = blockIdx.x * 8 + wave;
  const float* xr = x + (size_t)row * DD;
  float s = 0.f;
#pragma unroll 4
  for (int i = lane; i < DD; i += 32) { float v = xr[i]; s += v * v; }
  s = wsum(s);
  if (lane == 0) rnorm[row] = 1.0f / fmaxf(sqrtf(s), 1e-8f);
}

// ---------------- 2) s = sum_i x_i * rnorm_i (deterministic two-stage) ------
__global__ void k_colsum_partial(const float* __restrict__ x,
                                 const float* __restrict__ rnorm,
                                 float* __restrict__ partial) {
  const int t = threadIdx.x, bl = blockIdx.x;
  const int row0 = bl * 128;
  float a0 = 0.f, a1 = 0.f, a2 = 0.f, a3 = 0.f;
  for (int r = 0; r < 128; ++r) {
    const float rn = rnorm[row0 + r];
    const float* xr = x + (size_t)(row0 + r) * DD;
    a0 += xr[t]       * rn;
    a1 += xr[t + 256] * rn;
    a2 += xr[t + 512] * rn;
    a3 += xr[t + 768] * rn;
  }
  float* p = partial + (size_t)bl * DD;
  p[t] = a0; p[t + 256] = a1; p[t + 512] = a2; p[t + 768] = a3;
}

__global__ void k_colsum_reduce(const float* __restrict__ partial,
                                float* __restrict__ svec) {
  const int c = blockIdx.x * 256 + threadIdx.x;
  float s = 0.f;
#pragma unroll 8
  for (int b = 0; b < 64; ++b) s += partial[(size_t)b * DD + c];
  svec[c] = s;
}

// ---------------- 3) WbfT[c][d] = bf16(W[d][c]), zero-padded to 1024 cols ---
__global__ void k_wconvert(const float* __restrict__ W, __bf16* __restrict__ WbfT) {
  const int i = blockIdx.x * 256 + threadIdx.x;   // over 1024*1024
  const int d = i & (DD - 1);
  const int c = i >> 10;
  const float v = (c < CC) ? W[(size_t)d * CC + c] : 0.0f;
  WbfT[(size_t)c * DD + d] = (__bf16)v;
}

// ---------------- 4) weight[i] = (x_i . s) * rnorm_i / B --------------------
__global__ void k_weight(const float* __restrict__ x, const float* __restrict__ svec,
                         const float* __restrict__ rnorm, float* __restrict__ weight) {
  const int wave = threadIdx.x >> 5, lane = threadIdx.x & 31;
  const int row  = blockIdx.x * 8 + wave;
  const float* xr = x + (size_t)row * DD;
  float s = 0.f;
#pragma unroll 4
  for (int i = lane; i < DD; i += 32) s += xr[i] * svec[i];
  s = wsum(s);
  if (lane == 0) weight[row] = s * rnorm[row] * (1.0f / BB);
}

// ---------------- 5) fused bf16-WMMA GEMM + log-softmax entropy -------------
// 32 rows per workgroup: each B fragment feeds two A fragments (halves L2
// traffic on W, 3 b128 loads per WMMA instead of 4).
#define MROWS   32
#define ASTRIDE 1032                 // bf16 elems per A row in LDS (bank shift pad)
#define LSTRIDE 1028                 // f32 elems per logits row in LDS
#define A_BYTES (MROWS * ASTRIDE * 2)
#define L_BYTES (MROWS * LSTRIDE * 4)
#define SMEM_BYTES (A_BYTES + L_BYTES)

__global__ void __launch_bounds__(256)
k_main(const float* __restrict__ x, const __bf16* __restrict__ WbfT,
       const float* __restrict__ bias, const float* __restrict__ weight,
       float* __restrict__ out) {
  extern __shared__ char smem[];
  __bf16* Abf  = (__bf16*)smem;
  float*  Llog = (float*)(smem + A_BYTES);

  const int tid  = threadIdx.x;
  const int wave = tid >> 5, lane = tid & 31;
  const int row0 = blockIdx.x * MROWS;

  // stage 32x1024 x-tile into LDS as bf16 (coalesced along columns)
  for (int idx = tid; idx < MROWS * DD; idx += 256) {
    const int r = idx >> 10, c = idx & (DD - 1);
    Abf[r * ASTRIDE + c] = (__bf16)x[(size_t)(row0 + r) * DD + c];
  }
  __syncthreads();

  const int g  = lane >> 4;          // half-wave group
  const int hl = lane & 15;
  const __bf16* Arow0 = Abf + hl * ASTRIDE;              // rows 0..15
  const __bf16* Arow1 = Abf + (16 + hl) * ASTRIDE;       // rows 16..31

  // each wave owns 8 of the 64 N-tiles
  for (int nt = wave; nt < CPAD / 16; nt += 8) {
    const int ccol = nt * 16 + hl;                       // B fragment: lane -> column N
    const __bf16* Bp = WbfT + (size_t)ccol * DD + 16 * g;
    f32x8 acc0 = {0.f, 0.f, 0.f, 0.f, 0.f, 0.f, 0.f, 0.f};
    f32x8 acc1 = {0.f, 0.f, 0.f, 0.f, 0.f, 0.f, 0.f, 0.f};
#pragma unroll 2
    for (int k0 = 0; k0 < DD; k0 += 32) {
      // B 32x16 bf16 frag: element e -> K = k0 + 16g + e (contiguous in WbfT row)
      const bf16x8* bp = (const bf16x8*)(Bp + k0);
      bf16x8 blo = bp[0];
      bf16x8 bhi = bp[1];
      bf16x16 bfrag = __builtin_shufflevector(blo, bhi,
          0, 1, 2, 3, 4, 5, 6, 7, 8, 9, 10, 11, 12, 13, 14, 15);
      // A 16x32 bf16 frags: element e -> K = k0 + 16g + e
      const bf16x8* a0p = (const bf16x8*)(Arow0 + k0);
      bf16x16 af0 = __builtin_shufflevector(a0p[2 * g], a0p[2 * g + 1],
          0, 1, 2, 3, 4, 5, 6, 7, 8, 9, 10, 11, 12, 13, 14, 15);
      const bf16x8* a1p = (const bf16x8*)(Arow1 + k0);
      bf16x16 af1 = __builtin_shufflevector(a1p[2 * g], a1p[2 * g + 1],
          0, 1, 2, 3, 4, 5, 6, 7, 8, 9, 10, 11, 12, 13, 14, 15);
      acc0 = __builtin_amdgcn_wmma_f32_16x16x32_bf16(
          false, af0, false, bfrag, (short)0, acc0, false, false);
      acc1 = __builtin_amdgcn_wmma_f32_16x16x32_bf16(
          false, af1, false, bfrag, (short)0, acc1, false, false);
    }
    // C/D layout: lane -> col N = hl, VGPR r -> row M = r + 8*g
    float* lcol0 = Llog + (size_t)(8 * g) * LSTRIDE + nt * 16 + hl;
    float* lcol1 = lcol0 + (size_t)16 * LSTRIDE;
#pragma unroll
    for (int r = 0; r < 8; ++r) lcol0[r * LSTRIDE] = acc0[r];
#pragma unroll
    for (int r = 0; r < 8; ++r) lcol1[r * LSTRIDE] = acc1[r];
  }
  __syncthreads();

  // log-softmax negative entropy per row; 4 rows per wave
#pragma unroll
  for (int rr = 0; rr < 4; ++rr) {
    const int r = wave * 4 + rr;
    const float* lr = Llog + (size_t)r * LSTRIDE;
    float m = -3.4e38f;
    for (int c = lane; c < CC; c += 32) m = fmaxf(m, lr[c] + bias[c]);
    m = wmaxr(m);
    float s1 = 0.f, s2 = 0.f;
    for (int c = lane; c < CC; c += 32) {
      const float l = lr[c] + bias[c] - m;
      const float e = __expf(l);
      s1 += e;
      s2 += e * l;
    }
    s1 = wsum(s1);
    s2 = wsum(s2);
    if (lane == 0) {
      const float loss = s2 / s1 - __logf(s1);   // = sum p*logp
      out[row0 + r] = loss * weight[row0 + r];
    }
  }
}

// ---------------------------------------------------------------------------
extern "C" void kernel_launch(void* const* d_in, const int* in_sizes, int n_in,
                              void* d_out, int out_size, void* d_ws, size_t ws_size,
                              hipStream_t stream) {
  (void)in_sizes; (void)n_in; (void)out_size; (void)ws_size;
  const float* x    = (const float*)d_in[0];
  const float* W    = (const float*)d_in[1];
  const float* bvec = (const float*)d_in[2];
  float* out = (float*)d_out;

  // workspace layout (floats), then 16B-aligned bf16 region
  float* rnorm   = (float*)d_ws;            // 8192
  float* weight  = rnorm + BB;              // 8192
  float* svec    = weight + BB;             // 1024
  float* partial = svec + DD;               // 64*1024
  __bf16* WbfT   = (__bf16*)((char*)d_ws +
                   (size_t)(BB + BB + DD + 64 * DD) * sizeof(float)); // 2 MB, 16B aligned

  k_rnorm<<<BB / 8, 256, 0, stream>>>(x, rnorm);
  k_colsum_partial<<<64, 256, 0, stream>>>(x, rnorm, partial);
  k_colsum_reduce<<<DD / 256, 256, 0, stream>>>(partial, svec);
  k_wconvert<<<(DD * CPAD) / 256, 256, 0, stream>>>(W, WbfT);
  k_weight<<<BB / 8, 256, 0, stream>>>(x, svec, rnorm, weight);
  k_main<<<BB / MROWS, 256, SMEM_BYTES, stream>>>(x, WbfT, bvec, weight, out);
}
